// VenomSpmm_54090818126180
// MI455X (gfx1250) — compile-verified, matrix operand order checked by reading, not run on hardware
//
#include <hip/hip_runtime.h>

// ---- CDNA5 (gfx1250) wave32 WMMA types ----
typedef __attribute__((ext_vector_type(16))) __bf16         v16bf;
typedef __attribute__((ext_vector_type(2)))  __bf16         v2bf;
typedef __attribute__((ext_vector_type(8)))  float          v8f;
typedef __attribute__((ext_vector_type(8)))  unsigned short u16x8;
typedef __attribute__((ext_vector_type(16))) unsigned short u16x16;

#define TILE_M   256
#define TILE_N   128
#define KC       32              // compressed-k per LDS stage (bf16 elems)
#define KPAD     8               // pad to decorrelate LDS banks
#define LDK      (KC + KPAD)     // 40 halves = 80 B row stride
#define STAGES   32              // K_eff (1024) / KC
#define THREADS  256
#define N_DIM    4096
#define K_DIM    4096

// native f32->bf16 (RTNE) pair pack -> v_cvt_pk_bf16_f32 (confirmed in asm)
__device__ __forceinline__ unsigned int pack2_bf16(float a, float b) {
    v2bf p;
    p[0] = (__bf16)a;
    p[1] = (__bf16)b;
    return __builtin_bit_cast(unsigned int, p);
}

__device__ __forceinline__ v16bf load_frag(const unsigned short* p) {
    // ISA 16-bit A/B layout: lane holds k = 8*hi .. 8*hi+7 and 8*hi+16 .. 8*hi+23
    u16x8 lo = *(const u16x8*)(p);
    u16x8 hh = *(const u16x8*)(p + 16);
    u16x16 full = __builtin_shufflevector(lo, hh,
        0, 1, 2, 3, 4, 5, 6, 7, 8, 9, 10, 11, 12, 13, 14, 15);
    return __builtin_bit_cast(v16bf, full);
}

__global__ __launch_bounds__(THREADS)
void venom_spmm_bf16_wmma(const float* __restrict__ x,
                          const float* __restrict__ w,
                          const float* __restrict__ bias,
                          float* __restrict__ out) {
    // double-buffered stages: 2*(A 20KB + B 10KB) = 60KB of the 320KB WGP LDS
    __shared__ __attribute__((aligned(16))) unsigned short As[2][TILE_M * LDK];
    __shared__ __attribute__((aligned(16))) unsigned short Bs[2][TILE_N * LDK];

    const int tid  = threadIdx.x;
    const int lane = tid & 31;
    const int wave = tid >> 5;
    const int hi   = lane >> 4;   // half-wave select
    const int l16  = lane & 15;

    // 8 waves in a 4(M) x 2(N) grid; each wave owns a 64x64 output tile
    const int waveM = (wave >> 1) * 64;   // 0,64,128,192
    const int waveN = (wave & 1) * 64;    // 0,64

    const size_t m0 = (size_t)blockIdx.y * TILE_M;   // N-major grid => L2 reuse of x band
    const size_t n0 = (size_t)blockIdx.x * TILE_N;

    // per-thread staging slot (fixed across stages)
    const int srow = tid >> 4;          // base row; strides by 16
    const int sq   = tid & 15;          // quad column within stage (orig k = 8*sq)

    v8f acc[4][4] = {};   // sixteen 16x16 f32 accumulators per wave

    // ---- prologue: stage 0 -> LDS buffer 0 ----
    {
        size_t ko = (size_t)(8 * sq);
        #pragma unroll
        for (int it = 0; it < 16; ++it) {
            int row = srow + 16 * it;
            size_t base = (m0 + row) * (size_t)K_DIM + ko;
            ((unsigned int*)&As[0][0])[row * (LDK / 2) + sq] =
                pack2_bf16(x[base], x[base + 2]);
        }
        #pragma unroll
        for (int it = 0; it < 8; ++it) {
            int row = srow + 16 * it;
            size_t base = (n0 + row) * (size_t)K_DIM + ko;
            ((unsigned int*)&Bs[0][0])[row * (LDK / 2) + sq] =
                pack2_bf16(w[base], w[base + 2]);
        }
    }

    for (int s = 0; s < STAGES; ++s) {
        const int cur = s & 1;
        __syncthreads();   // LDS[cur] ready; prior reads of LDS[cur^1] retired

        // ---- issue next stage's global loads (in flight during wmma) ----
        float a0[16], a2[16], b0[8], b2[8];
        const bool pre = (s + 1 < STAGES);
        if (pre) {
            size_t ko = (size_t)(s + 1) * 128 + 8 * sq;   // mask keeps k%8 in {0,2}
            #pragma unroll
            for (int it = 0; it < 16; ++it) {
                int row = srow + 16 * it;
                size_t base = (m0 + row) * (size_t)K_DIM + ko;
                a0[it] = x[base];
                a2[it] = x[base + 2];
            }
            #pragma unroll
            for (int it = 0; it < 8; ++it) {
                int row = srow + 16 * it;
                size_t base = (n0 + row) * (size_t)K_DIM + ko;
                b0[it] = w[base];
                b2[it] = w[base + 2];
            }
        }

        // ---- fragments from LDS[cur] (b128 loads, conflict-free pad) ----
        v16bf afr[4], bfr[4];
        #pragma unroll
        for (int sm = 0; sm < 4; ++sm)
            afr[sm] = load_frag(&As[cur][(waveM + sm * 16 + l16) * LDK + 8 * hi]);
        #pragma unroll
        for (int sn = 0; sn < 4; ++sn)
            bfr[sn] = load_frag(&Bs[cur][(waveN + sn * 16 + l16) * LDK + 8 * hi]);

        // ---- 16x v_wmma_f32_16x16x32_bf16, f32 accumulate ----
        #pragma unroll
        for (int sm = 0; sm < 4; ++sm)
            #pragma unroll
            for (int sn = 0; sn < 4; ++sn)
                acc[sm][sn] = __builtin_amdgcn_wmma_f32_16x16x32_bf16(
                    false, afr[sm], false, bfr[sn],
                    (short)0, acc[sm][sn], false, false);

        // ---- convert + stash next stage into LDS[cur^1] ----
        if (pre) {
            #pragma unroll
            for (int it = 0; it < 16; ++it) {
                int row = srow + 16 * it;
                ((unsigned int*)&As[cur ^ 1][0])[row * (LDK / 2) + sq] =
                    pack2_bf16(a0[it], a2[it]);
            }
            #pragma unroll
            for (int it = 0; it < 8; ++it) {
                int row = srow + 16 * it;
                ((unsigned int*)&Bs[cur ^ 1][0])[row * (LDK / 2) + sq] =
                    pack2_bf16(b0[it], b2[it]);
            }
        }
    }

    // ---- epilogue: bias add + f32 store.
    // C/D layout: VGPR i -> row (i + 8*hi), col = lane&15 ----
    #pragma unroll
    for (int sm = 0; sm < 4; ++sm) {
        #pragma unroll
        for (int sn = 0; sn < 4; ++sn) {
            size_t col = n0 + waveN + sn * 16 + l16;
            float  bv  = bias[col];
            size_t rbase = m0 + waveM + sm * 16 + 8 * hi;
            #pragma unroll
            for (int i = 0; i < 8; ++i)
                out[(rbase + i) * (size_t)N_DIM + col] = acc[sm][sn][i] + bv;
        }
    }
}

extern "C" void kernel_launch(void* const* d_in, const int* in_sizes, int n_in,
                              void* d_out, int out_size, void* d_ws, size_t ws_size,
                              hipStream_t stream) {
    (void)in_sizes; (void)n_in; (void)d_ws; (void)ws_size; (void)out_size;
    const float* x    = (const float*)d_in[0];   // [4,4096,4096] f32
    const float* wgt  = (const float*)d_in[1];   // [4096,4096]  f32
    const float* bias = (const float*)d_in[2];   // [4096]       f32
    float* out = (float*)d_out;                  // [4,4096,4096] f32

    dim3 grid(N_DIM / TILE_N, (4 * 4096) / TILE_M);  // (32, 64), N-major
    venom_spmm_bf16_wmma<<<grid, THREADS, 0, stream>>>(x, wgt, bias, out);
}
